// ComplexDecoder_61890478735545
// MI455X (gfx1250) — matrix operand order repaired and checked by reading, not compiled, not run on hardware
//
#include <hip/hip_runtime.h>
#include <hip/hip_bf16.h>

// ---------------------------------------------------------------------------
// Types for CDNA5 WMMA (wave32, 16x16x32 bf16 -> f32)
// ---------------------------------------------------------------------------
typedef __bf16 bf16_t;
typedef __attribute__((ext_vector_type(16))) __bf16 v16bf;
typedef __attribute__((ext_vector_type(8)))  float  v8f;

__device__ inline bf16_t f2bf(float f) {
    union { float f; unsigned u; } a; a.f = f;
    unsigned u = a.u;
    unsigned r = (u + 0x7FFFu + ((u >> 16) & 1u)) >> 16;   // round-to-nearest-even
    union { unsigned short s; bf16_t b; } c; c.s = (unsigned short)r;
    return c.b;
}

// ---------------------------------------------------------------------------
// Generic doubled-real GEMM parameters
//   MA: 0 = packed bf16 weight matrix [M][K] row major
//       1 = A[i][c2] = q[z][c2][i]                (attention q^T)
//       2 = doubled-complex v: [[vr,-vi],[vi,vr]] (attention v)
//   MB: 0 = implicit im2col of planar activation [B][C2in][H][W]
//       1 = doubled-complex k: cols [kr;-ki | ki;kr]
//       2 = B[k][i] = w2[z][i][k]                 (scores transpose)
//   out_mode: 0 = conv layout out[b][m][hw] (n = b*HWo + hw)
//             1 = per-batch plane out[z][m][n]
// ---------------------------------------------------------------------------
struct GP {
    int mode_a, mode_b, out_mode;
    int M, N, K;
    const bf16_t* Wd;
    const float*  Aact;
    const float*  Bact;
    int C2in, H, W, ksize, pad, kk2;   // conv geometry for MB==0
    int HW;                            // spatial count for attention modes
    int Cc;                            // complex channel count for attention
    const float* br; const float* bi; int Co; int has_bias;
    const float* resid;
    float scale;
    float* out;
    int HWo;
};

#define BM 128
#define BN 128
#define BK 32

template <int MA, int MB>
__launch_bounds__(256)
__global__ void gemm_wmma_k(GP p) {
    __shared__ __align__(16) bf16_t As[BM][BK];   // [m][k], k contiguous
    __shared__ __align__(16) bf16_t Bs[BN][BK];   // [n][k], k contiguous

    const int tid  = threadIdx.x;
    const int lane = tid & 31;
    const int wave = tid >> 5;
    const int wm   = wave & 3;         // 4 waves along M (32 rows each)
    const int wn   = wave >> 2;        // 2 waves along N (64 cols each)
    const int m0   = blockIdx.y * BM;
    const int n0   = blockIdx.x * BN;
    const int z    = blockIdx.z;
    const int half = lane >> 4;
    const int lr   = lane & 15;

    v8f acc[2][4];
#pragma unroll
    for (int a = 0; a < 2; a++)
#pragma unroll
        for (int b = 0; b < 4; b++)
#pragma unroll
            for (int j = 0; j < 8; j++) acc[a][b][j] = 0.0f;

    for (int k0 = 0; k0 < p.K; k0 += BK) {
        // ---- stage A tile ------------------------------------------------
        if (MA == 0) {
            // K is always a multiple of BK; vectorized 4 x bf16 copies.
            for (int idx = tid; idx < BM * BK / 4; idx += 256) {
                const int ml = idx >> 3;
                const int kg = (idx & 7) * 4;
                const int m = m0 + ml;
                uint2 v = make_uint2(0u, 0u);
                if (m < p.M) {
                    const bf16_t* src = &p.Wd[(size_t)m * p.K + k0 + kg];
                    v = *(const uint2*)src;
                    __builtin_prefetch(src + BK, 0, 1);   // next k-slice
                }
                *(uint2*)&As[ml][kg] = v;
            }
        } else {
            for (int idx = tid; idx < BM * BK; idx += 256) {
                const int ml = idx >> 5, kl = idx & 31;
                const int m = m0 + ml, k = k0 + kl;
                float fv = 0.0f;
                if (m < p.M && k < p.K) {
                    if (MA == 1) {
                        fv = p.Aact[((size_t)z * 2 * p.Cc + k) * p.HW + m];
                    } else {
                        const int C = p.Cc, HW = p.HW;
                        const int c2 = m, k2 = k;
                        const int c  = (c2 < C) ? c2 : c2 - C;
                        const int hi = (k2 >= HW);
                        const int j  = hi ? k2 - HW : k2;
                        const int imagRow = (c2 >= C);
                        const int plane = (imagRow != hi) ? (C + c) : c;
                        const float s = (!imagRow && hi) ? -1.0f : 1.0f;
                        fv = s * p.Aact[((size_t)z * 2 * C + plane) * HW + j];
                    }
                }
                As[ml][kl] = f2bf(fv);
            }
        }
        // ---- stage B tile (consecutive threads -> consecutive n) ---------
        for (int idx = tid; idx < BN * BK; idx += 256) {
            const int kl = idx >> 7;          // BN == 128
            const int nl = idx & 127;
            const int n = n0 + nl, k = k0 + kl;
            float fv = 0.0f;
            if (n < p.N && k < p.K) {
                if (MB == 0) {
                    const int kk2 = p.kk2;
                    const int c2 = k / kk2, t = k - c2 * kk2;
                    const int r = t / p.ksize - p.pad;
                    const int s = t % p.ksize - p.pad;
                    const int HW = p.H * p.W;
                    const int b = n / HW, hw = n - b * HW;
                    const int y = hw / p.W, x = hw - y * p.W;
                    const int yy = y + r, xx = x + s;
                    if (yy >= 0 && yy < p.H && xx >= 0 && xx < p.W)
                        fv = p.Bact[((size_t)b * p.C2in + c2) * HW + yy * p.W + xx];
                } else if (MB == 1) {
                    const int C = p.Cc, HW = p.HW;
                    const int c2 = k;
                    const int hi = (n >= HW);
                    const int j  = hi ? n - HW : n;
                    int plane; float sgn = 1.0f;
                    if (c2 < C) plane = hi ? C + c2 : c2;
                    else { plane = hi ? c2 - C : c2; if (!hi) sgn = -1.0f; }
                    fv = sgn * p.Bact[((size_t)z * 2 * C + plane) * HW + j];
                } else {
                    fv = p.Bact[((size_t)z * p.HW + n) * (size_t)(2 * p.HW) + k];
                }
            }
            Bs[nl][kl] = f2bf(fv);
        }
        __syncthreads();

        // ---- fragment loads (ISA 16-bit A 16x32 / B 32x16 layouts) -------
        union Frag { uint4 q[2]; v16bf v; };
        Frag af[2], bfr[4];
#pragma unroll
        for (int ms = 0; ms < 2; ms++) {
            const bf16_t* ap = &As[wm * 32 + ms * 16 + lr][half ? 8 : 0];
            af[ms].q[0] = *(const uint4*)ap;         // K 0-7   / 8-15
            af[ms].q[1] = *(const uint4*)(ap + 16);  // K 16-23 / 24-31
        }
#pragma unroll
        for (int ns = 0; ns < 4; ns++) {
            const bf16_t* bp = &Bs[wn * 64 + ns * 16 + lr][half * 8];
            bfr[ns].q[0] = *(const uint4*)bp;        // K 0-7   / 8-15
            bfr[ns].q[1] = *(const uint4*)(bp + 16); // K 16-23 / 24-31
        }
#pragma unroll
        for (int ms = 0; ms < 2; ms++)
#pragma unroll
            for (int ns = 0; ns < 4; ns++)
                acc[ms][ns] = __builtin_amdgcn_wmma_f32_16x16x32_bf16(
                    false, af[ms].v, false, bfr[ns].v,
                    (short)0, acc[ms][ns], false, false);
        __syncthreads();
    }

    // ---- epilogue: scale + bias + residual + store -----------------------
#pragma unroll
    for (int ms = 0; ms < 2; ms++)
#pragma unroll
        for (int ns = 0; ns < 4; ns++)
#pragma unroll
            for (int j = 0; j < 8; j++) {
                const int m = m0 + wm * 32 + ms * 16 + j + (half ? 8 : 0);
                const int n = n0 + wn * 64 + ns * 16 + lr;
                if (m < p.M && n < p.N) {
                    float v = acc[ms][ns][j] * p.scale;
                    if (p.has_bias) v += (m < p.Co) ? p.br[m] : p.bi[m - p.Co];
                    size_t addr;
                    if (p.out_mode == 0) {
                        const int b = n / p.HWo, hw = n - b * p.HWo;
                        addr = ((size_t)b * p.M + m) * p.HWo + hw;
                    } else {
                        addr = ((size_t)z * p.M + m) * p.N + n;
                    }
                    if (p.resid) v += p.resid[addr];
                    p.out[addr] = v;
                }
            }
}

// ---------------------------------------------------------------------------
// Weight packing: complex (o,i,kh,kw) -> bf16 doubled-real [2o][2i*k^2]
// ---------------------------------------------------------------------------
__global__ void pack_w_k(const float* wr, const float* wi, bf16_t* dst,
                         int o, int i, int kk2) {
    const int total = 4 * o * i * kk2;
    const int idx = blockIdx.x * 256 + threadIdx.x;
    if (idx >= total) return;
    const int K = 2 * i * kk2;
    const int m = idx / K, k = idx - m * K;
    const int c2 = k / kk2, t = k - c2 * kk2;
    float v;
    if (m < o) {
        if (c2 < i) v =  wr[((size_t)m * i + c2) * kk2 + t];
        else        v = -wi[((size_t)m * i + (c2 - i)) * kk2 + t];
    } else {
        const int mm = m - o;
        if (c2 < i) v = wi[((size_t)mm * i + c2) * kk2 + t];
        else        v = wr[((size_t)mm * i + (c2 - i)) * kk2 + t];
    }
    dst[idx] = f2bf(v);
}

// ---------------------------------------------------------------------------
// ComplexGroupNorm (32 groups over interleaved r/i channels) + optional
// fused nonlin  x * sigmoid(|x|).  Planar storage [B][2C][HW].
// ---------------------------------------------------------------------------
__global__ void cgroupnorm_k(const float* in, float* out, const float* g,
                             const float* b, int C, int HW, int fuse) {
    const int grp = blockIdx.x, bz = blockIdx.y, tid = threadIdx.x;
    const int cpg = C / 32;                 // complex channels per group
    const float* base  = in  + (size_t)bz * 2 * C * HW;
    float*       obase = out + (size_t)bz * 2 * C * HW;
    const int nelem = cpg * HW * 2;
    float sum = 0.f, sq = 0.f;
    for (int idx = tid; idx < nelem; idx += 256) {
        const int cl = idx / (2 * HW);
        const int rem = idx - cl * 2 * HW;
        const int comp = rem / HW;
        const int hw = rem - comp * HW;
        const int plane = (comp ? C : 0) + grp * cpg + cl;
        const float v = base[(size_t)plane * HW + hw];
        sum += v; sq += v * v;
    }
    __shared__ float s1[256], s2[256];
    s1[tid] = sum; s2[tid] = sq; __syncthreads();
    for (int st = 128; st > 0; st >>= 1) {
        if (tid < st) { s1[tid] += s1[tid + st]; s2[tid] += s2[tid + st]; }
        __syncthreads();
    }
    const float mean = s1[0] / nelem;
    const float var  = s2[0] / nelem - mean * mean;
    const float rs   = rsqrtf(var + 1e-6f);
    for (int idx = tid; idx < cpg * HW; idx += 256) {
        const int cl = idx / HW, hw = idx - cl * HW;
        const int c = grp * cpg + cl;
        const float r  = base[(size_t)c * HW + hw];
        const float im = base[(size_t)(C + c) * HW + hw];
        float rn  = (r  - mean) * rs * g[2 * c]     + b[2 * c];
        float inn = (im - mean) * rs * g[2 * c + 1] + b[2 * c + 1];
        if (fuse) {
            const float mag = sqrtf(rn * rn + inn * inn);
            const float s = 1.0f / (1.0f + __expf(-mag));
            rn *= s; inn *= s;
        }
        obase[(size_t)c * HW + hw]       = rn;
        obase[(size_t)(C + c) * HW + hw] = inn;
    }
}

// ---------------------------------------------------------------------------
// ComplexSoftmax over magnitudes, phase preserved.  w layout [B][HW][2*HW]
// (real block then imag block per row).  One 256-thread block per row.
// ---------------------------------------------------------------------------
__global__ void csoftmax_k(float* w, int HW) {
    const int i = blockIdx.x, z = blockIdx.y, tid = threadIdx.x;
    float* base = w + ((size_t)z * HW + i) * (size_t)(2 * HW);
    float r[4], im[4], mag[4];
#pragma unroll
    for (int t = 0; t < 4; t++) {
        const int j = tid + t * 256;
        r[t] = base[j]; im[t] = base[HW + j];
        mag[t] = sqrtf(r[t] * r[t] + im[t] * im[t]);
    }
    __shared__ float red[256];
    float mx = fmaxf(fmaxf(mag[0], mag[1]), fmaxf(mag[2], mag[3]));
    red[tid] = mx; __syncthreads();
    for (int st = 128; st > 0; st >>= 1) {
        if (tid < st) red[tid] = fmaxf(red[tid], red[tid + st]);
        __syncthreads();
    }
    mx = red[0]; __syncthreads();
    float e[4], sum = 0.f;
#pragma unroll
    for (int t = 0; t < 4; t++) { e[t] = __expf(mag[t] - mx); sum += e[t]; }
    red[tid] = sum; __syncthreads();
    for (int st = 128; st > 0; st >>= 1) {
        if (tid < st) red[tid] += red[tid + st];
        __syncthreads();
    }
    const float inv = 1.0f / (red[0] + 1e-8f);
#pragma unroll
    for (int t = 0; t < 4; t++) {
        const float sm = e[t] * inv;
        float c, s;
        if (mag[t] > 0.f) { c = r[t] / mag[t]; s = im[t] / mag[t]; }
        else              { c = 1.f; s = 0.f; }
        const int j = tid + t * 256;
        base[j]      = sm * c;
        base[HW + j] = sm * s;
    }
}

// ---------------------------------------------------------------------------
// Nearest 2x upsample on planar activations: [BC][H][W] -> [BC][2H][2W]
// ---------------------------------------------------------------------------
__global__ void upsample2_k(const float* in, float* out, int BC, int Hin, int Win) {
    const size_t total = (size_t)BC * 4 * Hin * Win;
    for (size_t idx = (size_t)blockIdx.x * 256 + threadIdx.x; idx < total;
         idx += (size_t)gridDim.x * 256) {
        const size_t plane = idx / (size_t)(4 * Hin * Win);
        const int p = (int)(idx - plane * (size_t)(4 * Hin * Win));
        const int y = p / (2 * Win), x = p - y * (2 * Win);
        out[idx] = in[plane * (size_t)(Hin * Win) + (y >> 1) * Win + (x >> 1)];
    }
}

// z_real/z_imag -> planar [B][2C][HW]
__global__ void packz_k(const float* zr, const float* zi, float* out,
                        int B, int C, int HW) {
    const int total = B * 2 * C * HW;
    const int idx = blockIdx.x * 256 + threadIdx.x;
    if (idx >= total) return;
    const int b = idx / (2 * C * HW);
    const int rem = idx - b * 2 * C * HW;
    const int c2 = rem / HW, hw = rem - c2 * HW;
    out[idx] = (c2 < C) ? zr[((size_t)b * C + c2) * HW + hw]
                        : zi[((size_t)b * C + (c2 - C)) * HW + hw];
}

// planar [B][2][HW] -> output [B][HW][2]
__global__ void interleave_out_k(const float* in, float* out, int B, int HW) {
    const int total = B * 2 * HW;
    const int idx = blockIdx.x * 256 + threadIdx.x;
    if (idx >= total) return;
    const int b = idx / (2 * HW);
    const int rem = idx - b * 2 * HW;
    const int comp = rem / HW, hw = rem - comp * HW;
    out[((size_t)b * HW + hw) * 2 + comp] = in[idx];
}

// ---------------------------------------------------------------------------
// Host orchestration
// ---------------------------------------------------------------------------
static inline void gemm_go(hipStream_t s, const GP& p, int zdim) {
    dim3 g((p.N + BN - 1) / BN, (p.M + BM - 1) / BM, zdim);
    if (p.mode_a == 0)      gemm_wmma_k<0, 0><<<g, dim3(256), 0, s>>>(p);
    else if (p.mode_a == 1) gemm_wmma_k<1, 1><<<g, dim3(256), 0, s>>>(p);
    else                    gemm_wmma_k<2, 2><<<g, dim3(256), 0, s>>>(p);
}

extern "C" void kernel_launch(void* const* d_in, const int* in_sizes, int n_in,
                              void* d_out, int out_size, void* d_ws, size_t ws_size,
                              hipStream_t stream) {
    (void)in_sizes; (void)n_in; (void)out_size; (void)ws_size;
    auto F = [&](int i) { return (const float*)d_in[i]; };

    // --- workspace bump allocator ----------------------------------------
    char* ws = (char*)d_ws;
    size_t off = 0;
    auto alloc = [&](size_t bytes) -> void* {
        void* p = ws + off;
        off += (bytes + 255) & ~(size_t)255;
        return p;
    };
    const size_t ACT = (size_t)8 * 512 * 4096 * 4;  // 67 MB, max activation
    float* buf0 = (float*)alloc(ACT);
    float* buf1 = (float*)alloc(ACT);
    float* buf2 = (float*)alloc(ACT);
    float* wbuf = (float*)alloc((size_t)8 * 1024 * 2048 * 4);  // attn scores
    float* qb   = (float*)alloc((size_t)8 * 512 * 1024 * 4);
    float* kb   = (float*)alloc((size_t)8 * 512 * 1024 * 4);
    float* vb   = (float*)alloc((size_t)8 * 512 * 1024 * 4);

    // --- pack all complex conv weights to bf16 doubled-real form ---------
    struct CW { int idx, o, i, ks; };
    const CW cw[16] = {
        {2, 256, 16, 3},   // conv_in
        {8, 256, 256, 3},  {14, 256, 256, 3},                 // mid1 c1,c2
        {20, 256, 256, 1}, {24, 256, 256, 1},                 // q, k
        {28, 256, 256, 1}, {32, 256, 256, 1},                 // v, proj
        {38, 256, 256, 3}, {44, 256, 256, 3},                 // mid2
        {50, 256, 256, 3}, {56, 256, 256, 3},                 // up1
        {60, 256, 256, 3},                                    // up1_conv
        {66, 128, 256, 3}, {72, 128, 128, 3}, {76, 128, 256, 1}, // up0 c1,c2,nin
        {82, 1, 128, 3}                                       // conv_out
    };
    bf16_t* Wd[16];
    for (int t = 0; t < 16; t++) {
        const int kk2 = cw[t].ks * cw[t].ks;
        const int total = 4 * cw[t].o * cw[t].i * kk2;
        Wd[t] = (bf16_t*)alloc((size_t)total * sizeof(bf16_t));
        pack_w_k<<<(total + 255) / 256, 256, 0, stream>>>(
            F(cw[t].idx), F(cw[t].idx + 1), Wd[t], cw[t].o, cw[t].i, kk2);
    }

    // --- helpers ----------------------------------------------------------
    auto conv = [&](int wt, const float* src, float* dst, int H, int W,
                    int bias_idx, const float* resid) {
        GP p{};
        p.mode_a = 0; p.mode_b = 0; p.out_mode = 0;
        p.M = 2 * cw[wt].o;
        p.K = 2 * cw[wt].i * cw[wt].ks * cw[wt].ks;
        p.N = 8 * H * W;
        p.Wd = Wd[wt]; p.Bact = src;
        p.C2in = 2 * cw[wt].i; p.H = H; p.W = W;
        p.ksize = cw[wt].ks; p.pad = (cw[wt].ks == 3) ? 1 : 0;
        p.kk2 = cw[wt].ks * cw[wt].ks;
        p.br = F(bias_idx); p.bi = F(bias_idx + 1);
        p.Co = cw[wt].o; p.has_bias = 1;
        p.resid = resid; p.scale = 1.0f; p.out = dst; p.HWo = H * W;
        gemm_go(stream, p, 1);
    };
    auto gnorm = [&](const float* src, float* dst, int C, int HW, int gidx, int fuse) {
        cgroupnorm_k<<<dim3(32, 8), 256, 0, stream>>>(src, dst, F(gidx), F(gidx + 1),
                                                      C, HW, fuse);
    };

    // --- pipeline ---------------------------------------------------------
    // z -> planar [8][32][32][32]
    packz_k<<<(8 * 32 * 1024 + 255) / 256, 256, 0, stream>>>(F(0), F(1), buf0, 8, 16, 1024);
    // conv_in
    conv(0, buf0, buf1, 32, 32, 4, nullptr);

    // mid1 resblock
    gnorm(buf1, buf0, 256, 1024, 6, 1);
    conv(1, buf0, buf2, 32, 32, 10, nullptr);
    gnorm(buf2, buf0, 256, 1024, 12, 1);
    conv(2, buf0, buf2, 32, 32, 16, buf1);          // x = buf2

    // attention block
    gnorm(buf2, buf0, 256, 1024, 18, 0);            // no nonlin
    conv(3, buf0, qb, 32, 32, 22, nullptr);
    conv(4, buf0, kb, 32, 32, 26, nullptr);
    conv(5, buf0, vb, 32, 32, 30, nullptr);
    {   // scores: w2[b][i][0:HW]=real, [HW:2HW]=imag, scaled by 256^-0.5
        GP p{};
        p.mode_a = 1; p.mode_b = 1; p.out_mode = 1;
        p.M = 1024; p.N = 2048; p.K = 512;
        p.Aact = qb; p.Bact = kb; p.HW = 1024; p.Cc = 256;
        p.scale = 0.0625f; p.out = wbuf;
        gemm_go(stream, p, 8);
    }
    csoftmax_k<<<dim3(1024, 8), 256, 0, stream>>>(wbuf, 1024);
    {   // h_ = v . w
        GP p{};
        p.mode_a = 2; p.mode_b = 2; p.out_mode = 1;
        p.M = 512; p.N = 1024; p.K = 2048;
        p.Aact = vb; p.Bact = wbuf; p.HW = 1024; p.Cc = 256;
        p.scale = 1.0f; p.out = buf0;
        gemm_go(stream, p, 8);
    }
    conv(6, buf0, buf1, 32, 32, 34, buf2);          // proj + residual, x = buf1

    // mid2 resblock
    gnorm(buf1, buf0, 256, 1024, 36, 1);
    conv(7, buf0, buf2, 32, 32, 40, nullptr);
    gnorm(buf2, buf0, 256, 1024, 42, 1);
    conv(8, buf0, buf2, 32, 32, 46, buf1);          // x = buf2

    // up1 resblock
    gnorm(buf2, buf0, 256, 1024, 48, 1);
    conv(9, buf0, buf1, 32, 32, 52, nullptr);
    gnorm(buf1, buf0, 256, 1024, 54, 1);
    conv(10, buf0, buf1, 32, 32, 58, buf2);         // x = buf1 (32x32)

    // upsample + up1_conv (64x64)
    upsample2_k<<<4096, 256, 0, stream>>>(buf1, buf0, 8 * 512, 32, 32);
    conv(11, buf0, buf1, 64, 64, 62, nullptr);      // x = buf1

    // up0 resblock (256 -> 128, nin shortcut)
    gnorm(buf1, buf0, 256, 4096, 64, 1);
    conv(12, buf0, buf2, 64, 64, 68, nullptr);      // c1
    gnorm(buf2, buf0, 128, 4096, 70, 1);
    conv(14, buf1, buf2, 64, 64, 78, nullptr);      // nin(x) -> buf2
    conv(13, buf0, buf1, 64, 64, 74, buf2);         // c2 + nin residual, x = buf1

    // norm_out + nonlin, conv_out
    gnorm(buf1, buf0, 128, 4096, 80, 1);
    conv(15, buf0, buf2, 64, 64, 84, nullptr);      // [8][2][4096]

    interleave_out_k<<<(8 * 2 * 4096 + 255) / 256, 256, 0, stream>>>(
        buf2, (float*)d_out, 8, 4096);
}
